// HybridGATSAGE_39651138076873
// MI455X (gfx1250) — matrix operand order repaired and checked by reading, not compiled, not run on hardware
//
#include <hip/hip_runtime.h>
#include <hip/hip_bf16.h>

typedef __bf16 bf16;
typedef __attribute__((ext_vector_type(16))) __bf16 v16bf;
typedef __attribute__((ext_vector_type(8)))  float  v8f;

// ---------------- ordered-float encoding for atomic max ----------------
__device__ __forceinline__ unsigned f2ord(float f) {
  unsigned u = __float_as_uint(f);
  return (u & 0x80000000u) ? ~u : (u | 0x80000000u);
}
__device__ __forceinline__ float ord2f(unsigned u) {
  return (u & 0x80000000u) ? __uint_as_float(u & 0x7fffffffu) : __uint_as_float(~u);
}
#define NEG_INF_ORD 0x007FFFFFu   // f2ord(-inf)

// ---------------- small utility kernels ----------------
__global__ void fill_f(float* p, int n, float v) {
  int i = blockIdx.x * blockDim.x + threadIdx.x;
  if (i < n) p[i] = v;
}
__global__ void fill_u(unsigned* p, int n, unsigned v) {
  int i = blockIdx.x * blockDim.x + threadIdx.x;
  if (i < n) p[i] = v;
}
__global__ void relu_k(float* p, int n) {
  int i = blockIdx.x * blockDim.x + threadIdx.x;
  if (i < n) p[i] = fmaxf(p[i], 0.f);
}

// ---------------- fragment k-index maps (ISA 7.12.2, 16-bit operands) ------
__device__ __forceinline__ int ka_map(int e, int half) {
  int pair = e >> 1, off = e & 1;
  return ((pair < 4) ? (2 * pair) : (8 + 2 * pair)) + 8 * half + off;
}
__device__ __forceinline__ int kb_map(int e, int half) {
  int pair = e >> 1, off = e & 1;
  return ((pair < 4) ? (2 * pair) : (8 + 2 * pair)) + 8 * half + off;
}

// ---- A swizzle: A[M,K] f32 -> fragment-order bf16 (rows padded/clamped) ----
// out[(((rt*nkt)+kt)*32 + lane)*16 + e] = A[rt*16+(lane&15)][kt*32 + ka_map(e,lane>>4)]
__global__ void swizzle_A(const float* __restrict__ A, bf16* __restrict__ out,
                          int M, int Mpad, int K) {
  int idx = blockIdx.x * blockDim.x + threadIdx.x;
  if (idx >= Mpad * K) return;
  int nkt = K >> 5;
  int e = idx & 15;
  int lane = (idx >> 4) & 31;
  int rem = idx >> 9;
  int kt = rem % nkt;
  int rt = rem / nkt;
  int row = (rt << 4) + (lane & 15);
  if (row >= M) row = M - 1;            // clamped rows feed never-written outputs
  int k = (kt << 5) + ka_map(e, lane >> 4);
  out[idx] = (bf16)A[(size_t)row * K + k];
}

// ---- B (weight) swizzle: W[K,NC] f32 -> fragment-order bf16 ----
// out[(((ct*nkt)+kt)*32 + lane)*16 + e] = W[kt*32 + kb_map(e,lane>>4)][ct*16 + (lane&15)]
__global__ void swizzle_B(const float* __restrict__ W, bf16* __restrict__ out,
                          int K, int NC) {
  int idx = blockIdx.x * blockDim.x + threadIdx.x;
  if (idx >= K * NC) return;
  int nkt = K >> 5;
  int e = idx & 15;
  int lane = (idx >> 4) & 31;
  int rem = idx >> 9;
  int kt = rem % nkt;
  int ct = rem / nkt;
  int k = (kt << 5) + kb_map(e, lane >> 4);
  int col = (ct << 4) + (lane & 15);
  out[idx] = (bf16)W[(size_t)k * NC + col];
}

// ---------------- GAT1 dense: h = x[N,9] @ W1[9,1024] ----------------
__global__ __launch_bounds__(256) void gat1_lin(const float* __restrict__ x,
                                                const float* __restrict__ W1,
                                                float* __restrict__ h, int n) {
  int node = blockIdx.x;
  if (node >= n) return;
  __shared__ float xr[9];
  if (threadIdx.x < 9) xr[threadIdx.x] = x[(size_t)node * 9 + threadIdx.x];
  __syncthreads();
  for (int c = threadIdx.x; c < 1024; c += 256) {
    float acc = 0.f;
#pragma unroll
    for (int k = 0; k < 9; ++k) acc += xr[k] * W1[k * 1024 + c];
    h[(size_t)node * 1024 + c] = acc;
  }
}

// ---------------- attention scores ----------------
__global__ __launch_bounds__(256) void att_scores(const float* __restrict__ h,
                                                  const float* __restrict__ a_s,
                                                  const float* __restrict__ a_d,
                                                  float* __restrict__ asrc,
                                                  float* __restrict__ adst,
                                                  int n, int Hh, int C) {
  int gid = blockIdx.x * blockDim.x + threadIdx.x;
  int idx = gid >> 5;          // one wave32 per (node, head)
  int lane = threadIdx.x & 31;
  if (idx >= n * Hh) return;
  int node = idx / Hh, head = idx % Hh;
  const float* hp = h + (size_t)node * Hh * C + (size_t)head * C;
  const float* as = a_s + head * C;
  const float* ad = a_d + head * C;
  float ss = 0.f, sd = 0.f;
  for (int c = lane; c < C; c += 32) {
    float v = hp[c];
    ss += v * as[c];
    sd += v * ad[c];
  }
  for (int o = 16; o; o >>= 1) {
    ss += __shfl_down(ss, o, 32);
    sd += __shfl_down(sd, o, 32);
  }
  if (lane == 0) { asrc[idx] = ss; adst[idx] = sd; }
}

__device__ __forceinline__ void edge_nodes(int e, const int* src, const int* dst,
                                           int E_, int& s, int& d) {
  if (e < E_) { s = src[e]; d = dst[e]; }
  else        { s = e - E_; d = s; }        // implicit self-loops
}

// ---------------- edge softmax pass 1: segment max ----------------
__global__ void edge_max_k(const int* __restrict__ src, const int* __restrict__ dst,
                           int E_, int n, int Hh,
                           const float* __restrict__ asrc, const float* __restrict__ adst,
                           unsigned* __restrict__ m) {
  int e = blockIdx.x * blockDim.x + threadIdx.x;
  if (e >= E_ + n) return;
  int s, d; edge_nodes(e, src, dst, E_, s, d);
  for (int h = 0; h < Hh; ++h) {
    float x = asrc[s * Hh + h] + adst[d * Hh + h];
    x = (x > 0.f) ? x : 0.2f * x;           // leaky_relu 0.2
    atomicMax(&m[d * Hh + h], f2ord(x));
  }
}
// ---------------- edge softmax pass 2: segment sum of exp ----------------
__global__ void edge_sum_k(const int* __restrict__ src, const int* __restrict__ dst,
                           int E_, int n, int Hh,
                           const float* __restrict__ asrc, const float* __restrict__ adst,
                           const unsigned* __restrict__ m, float* __restrict__ ssum) {
  int e = blockIdx.x * blockDim.x + threadIdx.x;
  if (e >= E_ + n) return;
  int s, d; edge_nodes(e, src, dst, E_, s, d);
  for (int h = 0; h < Hh; ++h) {
    float x = asrc[s * Hh + h] + adst[d * Hh + h];
    x = (x > 0.f) ? x : 0.2f * x;
    atomicAdd(&ssum[d * Hh + h], __expf(x - ord2f(m[d * Hh + h])));
  }
}
// ---------------- edge softmax pass 3: weighted aggregation ----------------
__global__ __launch_bounds__(128) void edge_agg_k(const int* __restrict__ src,
                                                  const int* __restrict__ dst,
                                                  int E_, int n, int Hh, int Cshift,
                                                  const float* __restrict__ asrc,
                                                  const float* __restrict__ adst,
                                                  const unsigned* __restrict__ m,
                                                  const float* __restrict__ ssum,
                                                  const float* __restrict__ h,
                                                  float* __restrict__ out) {
  int e = blockIdx.x;
  if (e >= E_ + n) return;
  int s, d; edge_nodes(e, src, dst, E_, s, d);
  int D = Hh << Cshift;
  for (int f = threadIdx.x; f < D; f += 128) {
    int hh = f >> Cshift;
    float x = asrc[s * Hh + hh] + adst[d * Hh + hh];
    x = (x > 0.f) ? x : 0.2f * x;
    float alpha = __expf(x - ord2f(m[d * Hh + hh])) / ssum[d * Hh + hh];
    atomicAdd(&out[(size_t)d * D + f], alpha * h[(size_t)s * D + f]);
  }
}

// ---------------- SAGE helpers ----------------
__global__ void deg_k(const int* __restrict__ dst, float* __restrict__ deg, int E_) {
  int e = blockIdx.x * blockDim.x + threadIdx.x;
  if (e < E_) atomicAdd(&deg[dst[e]], 1.f);
}
__global__ __launch_bounds__(256) void scatter_add_rows(const int* __restrict__ src,
                                                        const int* __restrict__ dst,
                                                        const float* __restrict__ h,
                                                        float* __restrict__ out, int D) {
  int e = blockIdx.x;
  int s = src[e], d = dst[e];
  const float* hp = h + (size_t)s * D;
  float* op = out + (size_t)d * D;
  for (int c = threadIdx.x; c < D; c += 256) atomicAdd(&op[c], hp[c]);
}
__global__ __launch_bounds__(256) void row_scale(float* __restrict__ buf,
                                                 const float* __restrict__ deg, int D) {
  int row = blockIdx.x;
  float dv = deg[row]; if (dv < 1.f) dv = 1.f;
  float inv = 1.f / dv;
  for (int c = threadIdx.x; c < D; c += 256) buf[(size_t)row * D + c] *= inv;
}

// ---------------- bf16 WMMA GEMM on pre-swizzled operands ----------------
// Y[M,NC] = Aswz(Mpad,K) @ Bswz(K,NC), both fragment-order bf16.
// block = 128 threads = 4 waves -> 32 rows x 64 cols (2 row tiles per wave).
// Inner loop: 3x 32-byte (v16bf) global loads + 2x v_wmma. No LDS, no barriers.
__global__ __launch_bounds__(128) void gemm_bf16_wmma(
    const bf16* __restrict__ Aswz,
    const bf16* __restrict__ Bswz,
    float* __restrict__ Y, int ldy,
    const float* __restrict__ bias,
    int M, int K, int accumulate) {
  const int wave = threadIdx.x >> 5;
  const int lane = threadIdx.x & 31;
  const int nkt = K >> 5;
  const int rt0 = blockIdx.x * 2;
  const int ct  = blockIdx.y * 4 + wave;

  const bf16* ab0 = Aswz + (((size_t)rt0 * nkt) * 32 + lane) * 16;
  const bf16* ab1 = ab0 + (size_t)nkt * 512;
  const bf16* bb0 = Bswz + (((size_t)ct * nkt) * 32 + lane) * 16;

  v8f acc[2] = {{}, {}};
  for (int kt = 0; kt < nkt; ++kt) {
    v16bf a0 = *(const v16bf*)(ab0 + (size_t)kt * 512);
    v16bf a1 = *(const v16bf*)(ab1 + (size_t)kt * 512);
    v16bf bb = *(const v16bf*)(bb0 + (size_t)kt * 512);
    if (kt + 4 < nkt)
      __builtin_prefetch(bb0 + (size_t)(kt + 4) * 512, 0, 1);
    acc[0] = __builtin_amdgcn_wmma_f32_16x16x32_bf16(false, a0, false, bb,
                                                     (short)0, acc[0], false, false);
    acc[1] = __builtin_amdgcn_wmma_f32_16x16x32_bf16(false, a1, false, bb,
                                                     (short)0, acc[1], false, false);
  }
  // ---- writeback: lane -> col = lane&15, row = r + 8*(lane>>4) ----
  const int half = lane >> 4;
  const int col = blockIdx.y * 64 + wave * 16 + (lane & 15);
  const int row0 = blockIdx.x * 32;
#pragma unroll
  for (int t = 0; t < 2; ++t) {
#pragma unroll
    for (int r = 0; r < 8; ++r) {
      int row = row0 + t * 16 + r + 8 * half;
      if (row < M) {
        float vv = acc[t][r];
        if (bias) vv += bias[col];
        size_t idx = (size_t)row * ldy + col;
        if (accumulate) vv += Y[idx];
        Y[idx] = vv;
      }
    }
  }
}

// ---------------- LayerNorm (+optional pre-bias) + activation ----------------
__device__ __forceinline__ float block_reduce_sum_256(float val, float* sm) {
  int lane = threadIdx.x & 31, wv = threadIdx.x >> 5;
  for (int o = 16; o; o >>= 1) val += __shfl_down(val, o, 32);
  if (lane == 0) sm[wv] = val;
  __syncthreads();
  float t = (wv == 0 && lane < 8) ? sm[lane] : 0.f;
  if (wv == 0) {
    for (int o = 4; o; o >>= 1) t += __shfl_down(t, o, 32);
    if (lane == 0) sm[0] = t;
  }
  __syncthreads();
  float r = sm[0];
  __syncthreads();
  return r;
}

__global__ __launch_bounds__(256) void ln_act_k(const float* __restrict__ xin,
                                                const float* __restrict__ prebias,
                                                const float* __restrict__ gam,
                                                const float* __restrict__ bet,
                                                float* __restrict__ out, int D, int act) {
  __shared__ float sm[8];
  int row = blockIdx.x;
  const float* xr = xin + (size_t)row * D;
  float v[4];
  int cnt = 0;
  float s = 0.f;
  for (int c = threadIdx.x; c < D; c += 256) {
    float t = xr[c];
    if (prebias) t += prebias[c];
    v[cnt++] = t;
    s += t;
  }
  float mu = block_reduce_sum_256(s, sm) / (float)D;
  float vs = 0.f;
  for (int i = 0; i < cnt; ++i) { float dd = v[i] - mu; vs += dd * dd; }
  float rstd = rsqrtf(block_reduce_sum_256(vs, sm) / (float)D + 1e-5f);
  cnt = 0;
  for (int c = threadIdx.x; c < D; c += 256) {
    float t = (v[cnt++] - mu) * rstd * gam[c] + bet[c];
    if (act == 1)      t = (t > 0.f) ? t : (__expf(t) - 1.f);   // elu
    else if (act == 2) t = fmaxf(t, 0.f);                        // relu
    out[(size_t)row * D + c] = t;
  }
}

// ---------------- global mean/max pooling ----------------
__global__ __launch_bounds__(256) void pool_scatter(const float* __restrict__ h3,
                                                    const int* __restrict__ batch,
                                                    float* __restrict__ gsum,
                                                    unsigned* __restrict__ gmax,
                                                    float* __restrict__ cnt, int D) {
  int node = blockIdx.x;
  int b = batch[node];
  if (threadIdx.x == 0) atomicAdd(&cnt[b], 1.f);
  for (int c = threadIdx.x; c < D; c += 256) {
    float v = h3[(size_t)node * D + c];
    atomicAdd(&gsum[(size_t)b * D + c], v);
    atomicMax(&gmax[(size_t)b * D + c], f2ord(v));
  }
}
__global__ void pool_finalize(const float* __restrict__ gsum,
                              const unsigned* __restrict__ gmax,
                              const float* __restrict__ cnt,
                              float* __restrict__ g, int Bn, int D) {
  int idx = blockIdx.x * blockDim.x + threadIdx.x;
  if (idx >= Bn * D) return;
  int b = idx / D, c = idx % D;
  float cc = cnt[b]; if (cc < 1.f) cc = 1.f;
  g[(size_t)b * 2 * D + c]     = gsum[idx] / cc;
  g[(size_t)b * 2 * D + D + c] = ord2f(gmax[idx]);
}

// ---------------- final 64 -> 1 head with sigmoid ----------------
__global__ void final_head(const float* __restrict__ gin, const float* __restrict__ w,
                           const float* __restrict__ b, float* __restrict__ out, int Bn) {
  int i = blockIdx.x * blockDim.x + threadIdx.x;
  if (i >= Bn) return;
  float acc = b[0];
#pragma unroll
  for (int k = 0; k < 64; ++k) acc += gin[(size_t)i * 64 + k] * w[k];
  out[i] = 1.f / (1.f + __expf(-acc));
}

// =====================================================================
extern "C" void kernel_launch(void* const* d_in, const int* in_sizes, int n_in,
                              void* d_out, int out_size, void* d_ws, size_t ws_size,
                              hipStream_t stream) {
  const int N_ = in_sizes[0] / 9;
  const int E_ = in_sizes[1] / 2;
  const int Bn = out_size;          // 256 graphs
  const int Etot = E_ + N_;

  const float* x     = (const float*)d_in[0];
  const int*   ei    = (const int*)d_in[1];
  const int*   srcp  = ei;
  const int*   dstp  = ei + E_;
  const int*   batch = (const int*)d_in[2];
  const float* W1    = (const float*)d_in[3];
  const float* as1   = (const float*)d_in[4];
  const float* ad1   = (const float*)d_in[5];
  const float* b1    = (const float*)d_in[6];
  const float* ln1g  = (const float*)d_in[7];
  const float* ln1b  = (const float*)d_in[8];
  const float* Wl    = (const float*)d_in[9];
  const float* bl    = (const float*)d_in[10];
  const float* Wr    = (const float*)d_in[11];
  const float* ln2g  = (const float*)d_in[12];
  const float* ln2b  = (const float*)d_in[13];
  const float* W2    = (const float*)d_in[14];
  const float* as2   = (const float*)d_in[15];
  const float* ad2   = (const float*)d_in[16];
  const float* b2    = (const float*)d_in[17];
  const float* ln3g  = (const float*)d_in[18];
  const float* ln3b  = (const float*)d_in[19];
  const float* m1w   = (const float*)d_in[20];
  const float* m1b   = (const float*)d_in[21];
  const float* ml1g  = (const float*)d_in[22];
  const float* ml1b  = (const float*)d_in[23];
  const float* m2w   = (const float*)d_in[24];
  const float* m2b   = (const float*)d_in[25];
  const float* ml2g  = (const float*)d_in[26];
  const float* ml2b  = (const float*)d_in[27];
  const float* m3w   = (const float*)d_in[28];
  const float* m3b   = (const float*)d_in[29];
  const float* m4w   = (const float*)d_in[30];
  const float* m4b   = (const float*)d_in[31];

  auto cdiv = [](int a, int b) { return (a + b - 1) / b; };
  const int T = 256;
  const int Npad = cdiv(N_, 32) * 32;          // rows padded to 32 for GEMM tiles
  const int Bpad = cdiv(Bn, 32) * 32;

  // ---- workspace carve ----
  char* wp = (char*)d_ws;
  auto alloc = [&](size_t bytes) -> void* {
    void* p = (void*)wp;
    wp += (bytes + 255) & ~(size_t)255;
    return p;
  };
  float*    bufA  = (float*)alloc((size_t)N_ * 1024 * 4);   // feature buffer
  float*    bufB  = (float*)alloc((size_t)N_ * 1024 * 4);   // scratch buffer
  bf16*     Aswz  = (bf16*)alloc((size_t)Npad * 1024 * 2);  // swizzled activations
  float*    asrc  = (float*)alloc((size_t)N_ * 8 * 4);
  float*    adst  = (float*)alloc((size_t)N_ * 8 * 4);
  unsigned* mmax  = (unsigned*)alloc((size_t)N_ * 8 * 4);
  float*    ssum  = (float*)alloc((size_t)N_ * 8 * 4);
  float*    deg   = (float*)alloc((size_t)N_ * 4);
  float*    h2    = (float*)alloc((size_t)N_ * 128 * 4);
  float*    gsum  = (float*)alloc((size_t)Bn * 512 * 4);
  unsigned* gmax  = (unsigned*)alloc((size_t)Bn * 512 * 4);
  float*    cntb  = (float*)alloc((size_t)Bn * 4);
  float*    gcat  = (float*)alloc((size_t)Bn * 1024 * 4);
  float*    mlp1  = (float*)alloc((size_t)Bn * 256 * 4);
  float*    mlp2  = (float*)alloc((size_t)Bn * 128 * 4);
  float*    mlp3  = (float*)alloc((size_t)Bn * 64 * 4);
  bf16* Wl_sw = (bf16*)alloc((size_t)1024 * 128 * 2);
  bf16* Wr_sw = (bf16*)alloc((size_t)1024 * 128 * 2);
  bf16* W2_sw = (bf16*)alloc((size_t)128 * 512 * 2);
  bf16* m1_sw = (bf16*)alloc((size_t)1024 * 256 * 2);
  bf16* m2_sw = (bf16*)alloc((size_t)256 * 128 * 2);
  bf16* m3_sw = (bf16*)alloc((size_t)128 * 64 * 2);
  (void)ws_size; (void)n_in;

  // ---- weight swizzle to fragment-order bf16 ----
  swizzle_B<<<cdiv(1024 * 128, T), T, 0, stream>>>(Wl,  Wl_sw, 1024, 128);
  swizzle_B<<<cdiv(1024 * 128, T), T, 0, stream>>>(Wr,  Wr_sw, 1024, 128);
  swizzle_B<<<cdiv(128 * 512, T),  T, 0, stream>>>(W2,  W2_sw, 128, 512);
  swizzle_B<<<cdiv(1024 * 256, T), T, 0, stream>>>(m1w, m1_sw, 1024, 256);
  swizzle_B<<<cdiv(256 * 128, T),  T, 0, stream>>>(m2w, m2_sw, 256, 128);
  swizzle_B<<<cdiv(128 * 64, T),   T, 0, stream>>>(m3w, m3_sw, 128, 64);

  // ================= GAT1 =================
  gat1_lin<<<N_, T, 0, stream>>>(x, W1, bufA, N_);
  att_scores<<<cdiv(N_ * 8 * 32, T), T, 0, stream>>>(bufA, as1, ad1, asrc, adst, N_, 8, 128);
  fill_u<<<cdiv(N_ * 8, T), T, 0, stream>>>(mmax, N_ * 8, NEG_INF_ORD);
  fill_f<<<cdiv(N_ * 8, T), T, 0, stream>>>(ssum, N_ * 8, 0.f);
  fill_f<<<cdiv(N_ * 1024, T), T, 0, stream>>>(bufB, N_ * 1024, 0.f);
  edge_max_k<<<cdiv(Etot, T), T, 0, stream>>>(srcp, dstp, E_, N_, 8, asrc, adst, mmax);
  edge_sum_k<<<cdiv(Etot, T), T, 0, stream>>>(srcp, dstp, E_, N_, 8, asrc, adst, mmax, ssum);
  edge_agg_k<<<Etot, 128, 0, stream>>>(srcp, dstp, E_, N_, 8, 7, asrc, adst, mmax, ssum,
                                       bufA, bufB);
  ln_act_k<<<N_, T, 0, stream>>>(bufB, b1, ln1g, ln1b, bufA, 1024, 1);   // bufA = h

  // ================= SAGE =================
  fill_f<<<cdiv(N_, T), T, 0, stream>>>(deg, N_, 0.f);
  deg_k<<<cdiv(E_, T), T, 0, stream>>>(dstp, deg, E_);
  fill_f<<<cdiv(N_ * 1024, T), T, 0, stream>>>(bufB, N_ * 1024, 0.f);
  scatter_add_rows<<<E_, T, 0, stream>>>(srcp, dstp, bufA, bufB, 1024);
  row_scale<<<N_, T, 0, stream>>>(bufB, deg, 1024);                       // bufB = mean agg
  swizzle_A<<<cdiv(Npad * 1024, T), T, 0, stream>>>(bufB, Aswz, N_, Npad, 1024);
  gemm_bf16_wmma<<<dim3(Npad / 32, 2), 128, 0, stream>>>(Aswz, Wl_sw, h2, 128, bl,
                                                         N_, 1024, 0);
  swizzle_A<<<cdiv(Npad * 1024, T), T, 0, stream>>>(bufA, Aswz, N_, Npad, 1024);
  gemm_bf16_wmma<<<dim3(Npad / 32, 2), 128, 0, stream>>>(Aswz, Wr_sw, h2, 128,
                                                         (const float*)nullptr,
                                                         N_, 1024, 1);
  ln_act_k<<<N_, T, 0, stream>>>(h2, (const float*)nullptr, ln2g, ln2b, h2, 128, 1);

  // ================= GAT2 =================
  float* h3lin   = bufB;
  float* att2out = bufB + (size_t)N_ * 512;
  swizzle_A<<<cdiv(Npad * 128, T), T, 0, stream>>>(h2, Aswz, N_, Npad, 128);
  gemm_bf16_wmma<<<dim3(Npad / 32, 8), 128, 0, stream>>>(Aswz, W2_sw, h3lin, 512,
                                                         (const float*)nullptr,
                                                         N_, 128, 0);
  att_scores<<<cdiv(N_ * 8 * 32, T), T, 0, stream>>>(h3lin, as2, ad2, asrc, adst, N_, 8, 64);
  fill_u<<<cdiv(N_ * 8, T), T, 0, stream>>>(mmax, N_ * 8, NEG_INF_ORD);
  fill_f<<<cdiv(N_ * 8, T), T, 0, stream>>>(ssum, N_ * 8, 0.f);
  fill_f<<<cdiv(N_ * 512, T), T, 0, stream>>>(att2out, N_ * 512, 0.f);
  edge_max_k<<<cdiv(Etot, T), T, 0, stream>>>(srcp, dstp, E_, N_, 8, asrc, adst, mmax);
  edge_sum_k<<<cdiv(Etot, T), T, 0, stream>>>(srcp, dstp, E_, N_, 8, asrc, adst, mmax, ssum);
  edge_agg_k<<<Etot, 128, 0, stream>>>(srcp, dstp, E_, N_, 8, 6, asrc, adst, mmax, ssum,
                                       h3lin, att2out);
  ln_act_k<<<N_, T, 0, stream>>>(att2out, b2, ln3g, ln3b, bufA, 512, 1);  // bufA = h3

  // ================= pooling =================
  fill_f<<<cdiv(Bn * 512, T), T, 0, stream>>>(gsum, Bn * 512, 0.f);
  fill_u<<<cdiv(Bn * 512, T), T, 0, stream>>>(gmax, Bn * 512, NEG_INF_ORD);
  fill_f<<<cdiv(Bn, T), T, 0, stream>>>(cntb, Bn, 0.f);
  pool_scatter<<<N_, T, 0, stream>>>(bufA, batch, gsum, gmax, cntb, 512);
  pool_finalize<<<cdiv(Bn * 512, T), T, 0, stream>>>(gsum, gmax, cntb, gcat, Bn, 512);

  // ================= MLP =================
  swizzle_A<<<cdiv(Bpad * 1024, T), T, 0, stream>>>(gcat, Aswz, Bn, Bpad, 1024);
  gemm_bf16_wmma<<<dim3(Bpad / 32, 4), 128, 0, stream>>>(Aswz, m1_sw, mlp1, 256, m1b,
                                                         Bn, 1024, 0);
  ln_act_k<<<Bn, T, 0, stream>>>(mlp1, (const float*)nullptr, ml1g, ml1b, mlp1, 256, 2);
  swizzle_A<<<cdiv(Bpad * 256, T), T, 0, stream>>>(mlp1, Aswz, Bn, Bpad, 256);
  gemm_bf16_wmma<<<dim3(Bpad / 32, 2), 128, 0, stream>>>(Aswz, m2_sw, mlp2, 128, m2b,
                                                         Bn, 256, 0);
  ln_act_k<<<Bn, T, 0, stream>>>(mlp2, (const float*)nullptr, ml2g, ml2b, mlp2, 128, 2);
  swizzle_A<<<cdiv(Bpad * 128, T), T, 0, stream>>>(mlp2, Aswz, Bn, Bpad, 128);
  gemm_bf16_wmma<<<dim3(Bpad / 32, 1), 128, 0, stream>>>(Aswz, m3_sw, mlp3, 64, m3b,
                                                         Bn, 128, 0);
  relu_k<<<cdiv(Bn * 64, T), T, 0, stream>>>(mlp3, Bn * 64);
  final_head<<<cdiv(Bn, T), T, 0, stream>>>(mlp3, m4w, m4b, (float*)d_out, Bn);
}
